// ContextGatedDiffusionBlock_26834955665725
// MI455X (gfx1250) — matrix operand order repaired and checked by reading, not compiled, not run on hardware
//
#include <hip/hip_runtime.h>
#include <hip/hip_bf16.h>
#include <math.h>

#define HID   64
#define CTX   64
#define RANK  16
#define NNODE 1024
#define BTOT  16
#define TTOT  24
#define BT    (BTOT * TTOT)          // 384
#define BTN   ((size_t)BT * NNODE)   // 393216 tokens
#define NF    (BTN * HID)            // 25165824 elements
#define EPS   1e-5f

typedef __bf16 v16bf __attribute__((ext_vector_type(16)));
typedef __bf16 v8bf  __attribute__((ext_vector_type(8)));
typedef float  v8f   __attribute__((ext_vector_type(8)));

__device__ __forceinline__ float bf2f(__bf16 b) { return (float)b; }
__device__ __forceinline__ __bf16 f2bf(float f) { return (__bf16)f; }
__device__ __forceinline__ float gelu_exact(float x) {
    return 0.5f * x * (1.0f + erff(x * 0.70710678118654752f));
}
__device__ __forceinline__ float sigmoidf(float x) {
    return 1.0f / (1.0f + __expf(-x));
}
// 32-bit LDS byte offset for async-to-LDS DMA (ptrtoint of addrspace(3) ptr)
__device__ __forceinline__ unsigned lds_off32(const void* p) {
    return (unsigned)(unsigned long long)(__attribute__((address_space(3))) const void*)p;
}

// ---------------------------------------------------------------------------
// a_adp = softmax(relu(emb_src @ emb_dst^T), axis=-1), emitted as bf16
// ---------------------------------------------------------------------------
__global__ void __launch_bounds__(256) adj_adp_kernel(
    const float* __restrict__ es, const float* __restrict__ ed,
    __bf16* __restrict__ aout)
{
    __shared__ float red[256];
    __shared__ float esr[RANK];
    const int n = blockIdx.x, tid = threadIdx.x;
    if (tid < RANK) esr[tid] = es[n * RANK + tid];
    __syncthreads();

    float l[4];
#pragma unroll
    for (int j = 0; j < 4; ++j) {
        const int m = tid + j * 256;
        float d = 0.f;
#pragma unroll
        for (int r = 0; r < RANK; ++r) d += esr[r] * ed[m * RANK + r];
        l[j] = fmaxf(d, 0.f);
    }
    float mx = fmaxf(fmaxf(l[0], l[1]), fmaxf(l[2], l[3]));
    red[tid] = mx; __syncthreads();
    for (int s = 128; s > 0; s >>= 1) {
        if (tid < s) red[tid] = fmaxf(red[tid], red[tid + s]);
        __syncthreads();
    }
    mx = red[0]; __syncthreads();

    float e[4], sum = 0.f;
#pragma unroll
    for (int j = 0; j < 4; ++j) { e[j] = __expf(l[j] - mx); sum += e[j]; }
    red[tid] = sum; __syncthreads();
    for (int s = 128; s > 0; s >>= 1) {
        if (tid < s) red[tid] += red[tid + s];
        __syncthreads();
    }
    const float inv = 1.f / red[0];
#pragma unroll
    for (int j = 0; j < 4; ++j)
        aout[(size_t)n * NNODE + tid + j * 256] = f2bf(e[j] * inv);
}

// a_fwd -> bf16, a_bwd = a_fwd^T -> bf16
__global__ void __launch_bounds__(256) adj_conv_kernel(
    const float* __restrict__ af, __bf16* __restrict__ fw, __bf16* __restrict__ bw)
{
    const int n = blockIdx.x, tid = threadIdx.x;
#pragma unroll
    for (int j = 0; j < 4; ++j) {
        const int m = tid + j * 256;
        fw[(size_t)n * NNODE + m] = f2bf(af[(size_t)n * NNODE + m]);
        bw[(size_t)n * NNODE + m] = f2bf(af[(size_t)m * NNODE + n]);
    }
}

// ---------------------------------------------------------------------------
// Per-token MLPs. 256 threads = 4 tokens x 64 features.
// ---------------------------------------------------------------------------
__global__ void __launch_bounds__(256) prep_tokens_kernel(
    const float* __restrict__ x, const float* __restrict__ ctx,
    const float* __restrict__ Wg1, const float* __restrict__ bg1,
    const float* __restrict__ Wg2, const float* __restrict__ bg2,
    const float* __restrict__ Ws1, const float* __restrict__ bs1,
    const float* __restrict__ Ws2, const float* __restrict__ bs2,
    const float* __restrict__ Wsrc, const float* __restrict__ bsrc,
    const float* __restrict__ Wdst, const float* __restrict__ bdst,
    const float* __restrict__ Wself, const float* __restrict__ bself,
    float* __restrict__ mixed, float* __restrict__ gates3,
    float* __restrict__ scale, float* __restrict__ srcg, float* __restrict__ dstg)
{
    __shared__ float cs[4][HID], hs[4][HID], xs[4][HID], g0s[4];
    const int tid = threadIdx.x, g = tid >> 6, f = tid & 63;
    const size_t tk = (size_t)blockIdx.x * 4 + g;
    cs[g][f] = ctx[tk * CTX + f];
    xs[g][f] = x[tk * HID + f];
    __syncthreads();

    float acc = bg1[f];
#pragma unroll
    for (int i = 0; i < CTX; ++i) acc += cs[g][i] * Wg1[f * CTX + i];
    hs[g][f] = gelu_exact(acc);
    __syncthreads();

    if (f < 4) {
        float a = bg2[f];
#pragma unroll
        for (int i = 0; i < HID; ++i) a += hs[g][i] * Wg2[f * HID + i];
        const float gg = sigmoidf(a);
        if (f == 0) g0s[g] = gg; else gates3[tk * 3 + (f - 1)] = gg;
    }
    if (f == 4) {
        float a = bsrc[0];
#pragma unroll
        for (int i = 0; i < CTX; ++i) a += cs[g][i] * Wsrc[i];
        srcg[tk] = sigmoidf(a);
    }
    if (f == 5) {
        float a = bdst[0];
#pragma unroll
        for (int i = 0; i < CTX; ++i) a += cs[g][i] * Wdst[i];
        dstg[tk] = sigmoidf(a);
    }
    __syncthreads();

    acc = bs1[f];
#pragma unroll
    for (int i = 0; i < CTX; ++i) acc += cs[g][i] * Ws1[f * CTX + i];
    hs[g][f] = gelu_exact(acc);
    __syncthreads();

    float sa = bs2[f];
#pragma unroll
    for (int i = 0; i < HID; ++i) sa += hs[g][i] * Ws2[f * HID + i];
    scale[tk * HID + f] = sigmoidf(sa);

    float st = bself[f];
#pragma unroll
    for (int i = 0; i < HID; ++i) st += xs[g][i] * Wself[f * HID + i];
    mixed[tk * HID + f] = g0s[g] * st;
}

// ---------------------------------------------------------------------------
// X_t[bt][f][m] = bf16(x[bt][m][f] * gate[bt,m])  (gate==nullptr -> 1.0)
// LDS tile transpose, 64m x 64f per block. grid = (NNODE/64, BT)
// ---------------------------------------------------------------------------
__global__ void __launch_bounds__(256) transpose_gate_kernel(
    const float* __restrict__ x, const float* __restrict__ gate,
    __bf16* __restrict__ xt)
{
    __shared__ __bf16 tile[64][65];
    const int tid = threadIdx.x;
    const int bt = blockIdx.y, m0 = blockIdx.x * 64;

    {   // load: thread -> row r (m), 16 consecutive f
        const int r = tid >> 2, fc = (tid & 3) * 16;
        const size_t tok = (size_t)bt * NNODE + m0 + r;
        const float gv = gate ? gate[tok] : 1.0f;
        const float* src = x + tok * HID + fc;
#pragma unroll
        for (int j = 0; j < 16; ++j) tile[r][fc + j] = f2bf(src[j] * gv);
    }
    __syncthreads();
    {   // store: thread -> row fr (f), 16 consecutive m (contiguous in X_t)
        const int fr = tid >> 2, mc = (tid & 3) * 16;
        __bf16* dst = xt + ((size_t)bt * HID + fr) * NNODE + m0 + mc;
#pragma unroll
        for (int j = 0; j < 16; ++j) dst[j] = tile[mc + j][fr];
    }
}

// ---------------------------------------------------------------------------
// Diffusion GEMM (transposed-state layout):
//   Yt[bt][f][n] = sum_m A[n,m] * Xt[bt][f][m]
// grid = (NNODE/64, BT), block = 128 (4 waves; wave owns one 16-wide f-tile,
// 64 n-rows => 4 accumulators). K-step = 64 => 8 WMMAs per barrier pair.
// A tile (64x64 bf16, 8 KB) double-buffered via async DMA to LDS: the DMA for
// tile i+1 is issued before computing tile i and waited with
// s_wait_asynccnt<=4 (async loads complete in order).
// B fragments come straight from global (two 16B pieces/lane, transposed
// layout) and live in registers across the 4 n-subtiles.
// ---------------------------------------------------------------------------
#define KSTEP 64
__device__ __forceinline__ void issue_a_tile(
    const __bf16* __restrict__ A, int n0, int k0, __bf16* buf, int tid)
{
#pragma unroll
    for (int c = 0; c < 4; ++c) {
        const int chunk = tid * 4 + c;              // 0..511 16B chunks
        const int row = chunk >> 3, ch = chunk & 7;
        const unsigned long long gp =
            (unsigned long long)(A + (size_t)(n0 + row) * NNODE + (k0 + ch * 8));
        const unsigned la = lds_off32(buf + row * KSTEP + ch * 8);
        asm volatile("global_load_async_to_lds_b128 %0, %1, off"
                     :: "v"(la), "v"(gp) : "memory");
    }
}

__global__ void __launch_bounds__(128) diffuse_gemm_kernel(
    const __bf16* __restrict__ A, const __bf16* __restrict__ Xt,
    __bf16* __restrict__ Yt)
{
    __shared__ __align__(16) __bf16 As[2][64 * KSTEP];   // double-buffered A

    const int tid  = threadIdx.x;
    const int lane = tid & 31;
    const int wave = tid >> 5;
    const int n0   = blockIdx.x * 64;
    const size_t btbase = (size_t)blockIdx.y * ((size_t)HID * NNODE);

    const int r16 = lane & 15;
    const int hiK = lane >> 4;               // 0: lanes 0-15, 1: lanes 16-31
    // B operand source row: col f = wave*16 + r16, K contiguous in memory
    const __bf16* browBase = Xt + btbase + (size_t)(wave * 16 + r16) * NNODE;

    v8f acc[4];
#pragma unroll
    for (int t = 0; t < 4; ++t)
#pragma unroll
        for (int i = 0; i < 8; ++i) acc[t][i] = 0.f;

    // prologue: DMA tile 0
    issue_a_tile(A, n0, 0, &As[0][0], tid);

    for (int it = 0; it < NNODE / KSTEP; ++it) {
        const int k0 = it * KSTEP;
        __syncthreads();   // all waves done reading As[(it+1)&1] (iter it-1)

        // issue next tile (clamped dummy on last iter to keep counts uniform)
        const int knext = (k0 + KSTEP < NNODE) ? (k0 + KSTEP) : k0;
        issue_a_tile(A, n0, knext, &As[(it + 1) & 1][0], tid);

        // B fragments for this tile: 2 K-chunks of 32, two 16B pieces per lane (K 8h..8h+7, 16+8h..16+8h+7)
        v16bf b[2];
#pragma unroll
        for (int kc = 0; kc < 2; ++kc) {
            const __bf16* pb = browBase + k0 + kc * 32 + hiK * 8;
            const v8bf bLo = *(const v8bf*)pb;
            const v8bf bHi = *(const v8bf*)(pb + 16);
#pragma unroll
            for (int j = 0; j < 8; ++j) { b[kc][j] = bLo[j]; b[kc][8 + j] = bHi[j]; }
        }

        // wait for current tile's DMA (4 newly issued remain outstanding)
        asm volatile("s_wait_asynccnt 4" ::: "memory");
        __syncthreads();

        const __bf16* buf = &As[it & 1][0];
#pragma unroll
        for (int kc = 0; kc < 2; ++kc) {
#pragma unroll
            for (int t = 0; t < 4; ++t) {
                const __bf16* pa = buf + (t * 16 + r16) * KSTEP + kc * 32 + hiK * 8;
                const v8bf aLo = *(const v8bf*)pa;        // K 0-7  (or 8-15)
                const v8bf aHi = *(const v8bf*)(pa + 16); // K 16-23 (or 24-31)
                v16bf a;
#pragma unroll
                for (int j = 0; j < 8; ++j) { a[j] = aLo[j]; a[8 + j] = aHi[j]; }
                acc[t] = __builtin_amdgcn_wmma_f32_16x16x32_bf16(
                    false, a, false, b[kc], (short)0, acc[t], false, false);
            }
        }
    }
    // drain the dummy DMA before kernel end
    asm volatile("s_wait_asynccnt 0" ::: "memory");

    // Epilogue: C VGPR r -> (M = r + 8*hiK, N = r16); in transposed output the
    // 8 rows are 8 consecutive n -> one 16B store per sub-tile.
    const int fcol = wave * 16 + r16;
    const int M0 = hiK * 8;
#pragma unroll
    for (int t = 0; t < 4; ++t) {
        v8bf o;
#pragma unroll
        for (int r = 0; r < 8; ++r) o[r] = f2bf(acc[t][r]);
        *(v8bf*)(Yt + btbase + (size_t)fcol * NNODE + n0 + t * 16 + M0) = o;
    }
}

// ---------------------------------------------------------------------------
// mixed += gate * (y1 @ W[0]^T + b[0] + y2 @ W[1]^T + b[1])
// y1/y2 arrive in transposed layout [bt][f][n]; LDS tile-transpose first.
// grid = (NNODE/64, BT), block = 256.
// ---------------------------------------------------------------------------
__global__ void __launch_bounds__(256) accum_term_kernel(
    const __bf16* __restrict__ y1t, const __bf16* __restrict__ y2t,
    const float* __restrict__ W, const float* __restrict__ b,
    const float* __restrict__ gates3, int gcol, float* __restrict__ mixed)
{
    __shared__ float y1s[64][65];
    __shared__ float y2s[64][65];
    const int tid = threadIdx.x;
    const int bt = blockIdx.y, n0 = blockIdx.x * 64;
    const size_t btbase = (size_t)bt * HID * NNODE;

    {   // load coalesced along n, write LDS as [token][f]
        const int f = tid >> 2, c = (tid & 3) * 16;
        const __bf16* s1 = y1t + btbase + (size_t)f * NNODE + n0 + c;
        const __bf16* s2 = y2t + btbase + (size_t)f * NNODE + n0 + c;
#pragma unroll
        for (int j = 0; j < 16; ++j) {
            y1s[c + j][f] = bf2f(s1[j]);
            y2s[c + j][f] = bf2f(s2[j]);
        }
    }
    __syncthreads();

#pragma unroll
    for (int i = 0; i < 16; ++i) {
        const int idx = tid + i * 256;          // 0..4095
        const int tok = idx >> 6, o = idx & 63;
        float a = b[o] + b[HID + o];
        const float* W1 = W + (size_t)o * HID;
        const float* W2 = W + (size_t)HID * HID + (size_t)o * HID;
#pragma unroll
        for (int k = 0; k < HID; ++k) a += y1s[tok][k] * W1[k] + y2s[tok][k] * W2[k];
        const size_t tk = (size_t)bt * NNODE + n0 + tok;
        mixed[tk * HID + o] += gates3[tk * 3 + gcol] * a;
    }
}

// ---------------------------------------------------------------------------
// h = x + gelu(mixed * scale); layernorm over HID (wave32 shfl reductions)
// ---------------------------------------------------------------------------
__global__ void __launch_bounds__(256) final_ln_kernel(
    const float* __restrict__ x, const float* __restrict__ mixed,
    const float* __restrict__ scale, const float* __restrict__ gamma,
    const float* __restrict__ beta, float* __restrict__ out)
{
    const int tid = threadIdx.x, wave = tid >> 5, lane = tid & 31;
    const size_t base = ((size_t)blockIdx.x * 8 + wave) * HID;

    float h[2];
#pragma unroll
    for (int j = 0; j < 2; ++j) {
        const int f = lane + 32 * j;
        const float m = mixed[base + f] * scale[base + f];
        h[j] = x[base + f] + gelu_exact(m);
    }
    float s = h[0] + h[1];
    float s2 = h[0] * h[0] + h[1] * h[1];
#pragma unroll
    for (int off = 16; off > 0; off >>= 1) {
        s  += __shfl_xor(s,  off, 32);
        s2 += __shfl_xor(s2, off, 32);
    }
    const float mu  = s * (1.f / 64.f);
    const float var = s2 * (1.f / 64.f) - mu * mu;
    const float inv = rsqrtf(var + EPS);
#pragma unroll
    for (int j = 0; j < 2; ++j) {
        const int f = lane + 32 * j;
        out[base + f] = (h[j] - mu) * inv * gamma[f] + beta[f];
    }
}

// ---------------------------------------------------------------------------
extern "C" void kernel_launch(void* const* d_in, const int* in_sizes, int n_in,
                              void* d_out, int out_size, void* d_ws, size_t ws_size,
                              hipStream_t stream)
{
    const float* x       = (const float*)d_in[0];
    const float* a_fwd   = (const float*)d_in[1];
    const float* context = (const float*)d_in[2];
    const float* W_self  = (const float*)d_in[3];
    const float* b_self  = (const float*)d_in[4];
    const float* W_fwd   = (const float*)d_in[5];
    const float* b_fwd   = (const float*)d_in[6];
    const float* W_bwd   = (const float*)d_in[7];
    const float* b_bwd   = (const float*)d_in[8];
    const float* W_adp   = (const float*)d_in[9];
    const float* b_adp   = (const float*)d_in[10];
    const float* Wg1     = (const float*)d_in[11];
    const float* bg1     = (const float*)d_in[12];
    const float* Wg2     = (const float*)d_in[13];
    const float* bg2     = (const float*)d_in[14];
    const float* Ws1     = (const float*)d_in[15];
    const float* bs1     = (const float*)d_in[16];
    const float* Ws2     = (const float*)d_in[17];
    const float* bs2     = (const float*)d_in[18];
    const float* W_src   = (const float*)d_in[19];
    const float* b_src   = (const float*)d_in[20];
    const float* W_dst   = (const float*)d_in[21];
    const float* b_dst   = (const float*)d_in[22];
    const float* emb_src = (const float*)d_in[23];
    const float* emb_dst = (const float*)d_in[24];
    const float* gamma   = (const float*)d_in[25];
    const float* beta    = (const float*)d_in[26];
    float* out = (float*)d_out;

    char* p = (char*)d_ws;
    auto carve = [&](size_t bytes) -> void* {
        void* r = (void*)p;
        p += (bytes + 255) & ~(size_t)255;
        return r;
    };
    __bf16* aADP = (__bf16*)carve((size_t)NNODE * NNODE * 2);
    __bf16* aFWD = (__bf16*)carve((size_t)NNODE * NNODE * 2);
    __bf16* aBWD = (__bf16*)carve((size_t)NNODE * NNODE * 2);
    __bf16* xgT  = (__bf16*)carve(NF * 2);   // transposed [bt][f][m]
    __bf16* y1T  = (__bf16*)carve(NF * 2);   // transposed [bt][f][n]
    __bf16* y2T  = (__bf16*)carve(NF * 2);   // transposed [bt][f][n]
    float* mixed  = (float*)carve(NF * 4);
    float* scale  = (float*)carve(NF * 4);
    float* gates3 = (float*)carve(BTN * 3 * 4);
    float* srcg   = (float*)carve(BTN * 4);
    float* dstg   = (float*)carve(BTN * 4);

    adj_adp_kernel<<<NNODE, 256, 0, stream>>>(emb_src, emb_dst, aADP);
    adj_conv_kernel<<<NNODE, 256, 0, stream>>>(a_fwd, aFWD, aBWD);

    prep_tokens_kernel<<<(unsigned)(BTN / 4), 256, 0, stream>>>(
        x, context, Wg1, bg1, Wg2, bg2, Ws1, bs1, Ws2, bs2,
        W_src, b_src, W_dst, b_dst, W_self, b_self,
        mixed, gates3, scale, srcg, dstg);

    const __bf16* adjs[3]   = { aADP, aFWD, aBWD };
    const float*  gateIn[3] = { nullptr, dstg, srcg };
    const float*  Wd[3]     = { W_adp, W_fwd, W_bwd };
    const float*  bd[3]     = { b_adp, b_fwd, b_bwd };
    const int     gcol[3]   = { 2, 0, 1 };   // ws gate order: (g_fwd, g_bwd, g_adp)

    const dim3 tgrid(NNODE / 64, BT);
    for (int br = 0; br < 3; ++br) {
        transpose_gate_kernel<<<tgrid, 256, 0, stream>>>(x, gateIn[br], xgT);
        diffuse_gemm_kernel<<<tgrid, 128, 0, stream>>>(adjs[br], xgT, y1T);
        diffuse_gemm_kernel<<<tgrid, 128, 0, stream>>>(adjs[br], y1T, y2T);
        accum_term_kernel<<<tgrid, 256, 0, stream>>>(
            y1T, y2T, Wd[br], bd[br], gates3, gcol[br], mixed);
    }

    final_ln_kernel<<<(unsigned)(BTN / 8), 256, 0, stream>>>(
        x, mixed, scale, gamma, beta, out);

    (void)in_sizes; (void)n_in; (void)out_size; (void)ws_size;
}
